// MaskIoUHead_21260088115983
// MI455X (gfx1250) — compile-verified, matrix operand order checked by reading, not run on hardware
//
#include <hip/hip_runtime.h>
#include <math.h>

// ---------------------------------------------------------------------------
// Types for CDNA5 WMMA (wave32): bf16 A/B fragments (16 halves = 8 VGPRs),
// f32 accumulator (8 VGPRs).
// ---------------------------------------------------------------------------
typedef __bf16        bf16x16 __attribute__((ext_vector_type(16)));
typedef float         f32x8   __attribute__((ext_vector_type(8)));
typedef unsigned int  u32x4   __attribute__((ext_vector_type(4)));

union BfFrag {
    u32x4   u[2];   // two 16-byte global loads
    bf16x16 v;      // WMMA operand view
};

static __device__ __forceinline__ unsigned short f2bf(float f) {
    unsigned int u = __float_as_uint(f);
    u += 0x7FFFu + ((u >> 16) & 1u);          // round-to-nearest-even
    return (unsigned short)(u >> 16);
}
static __device__ __forceinline__ float bf2f(unsigned short h) {
    return __uint_as_float(((unsigned int)h) << 16);
}

#define HH 128
#define WW 128
#define HWSZ (HH * WW)
#define NB 128

// ---------------------------------------------------------------------------
// K0: f32 -> bf16 weight conversion (grid-stride, pure bandwidth)
// ---------------------------------------------------------------------------
__global__ void cvt_bf16_kernel(const float* __restrict__ src,
                                unsigned short* __restrict__ dst, int n) {
    for (int i = blockIdx.x * blockDim.x + threadIdx.x; i < n;
         i += gridDim.x * blockDim.x)
        dst[i] = f2bf(src[i]);
}

// ---------------------------------------------------------------------------
// K1: fused concat(x,mask) + 3x3 conv (9ch -> 2ch), pad=1, +bias
// one thread per (n,y,x); weights staged in LDS
// ---------------------------------------------------------------------------
__global__ void conv1_kernel(const float* __restrict__ x,
                             const float* __restrict__ mask,
                             const float* __restrict__ w1,
                             const float* __restrict__ b1,
                             float* __restrict__ h1) {
    __shared__ float sw[162];
    __shared__ float sb[2];
    int t = threadIdx.x;
    if (t < 162) sw[t] = w1[t];
    if (t < 2)   sb[t] = b1[t];
    __syncthreads();

    int idx = blockIdx.x * blockDim.x + t;
    if (idx >= NB * HWSZ) return;
    int xq = idx & (WW - 1);
    int yq = (idx >> 7) & (HH - 1);
    int n  = idx >> 14;

    float acc0 = sb[0], acc1 = sb[1];
#pragma unroll
    for (int ic = 0; ic < 9; ++ic) {
        const float* plane = (ic < 8) ? (x + ((size_t)n * 8 + ic) * HWSZ)
                                      : (mask + (size_t)n * HWSZ);
#pragma unroll
        for (int ky = 0; ky < 3; ++ky) {
            int yy = yq + ky - 1;
            if (yy < 0 || yy >= HH) continue;
#pragma unroll
            for (int kx = 0; kx < 3; ++kx) {
                int xx = xq + kx - 1;
                if (xx < 0 || xx >= WW) continue;
                float v = plane[yy * WW + xx];
                int wi = ic * 9 + ky * 3 + kx;         // (oc,ic,ky,kx): oc*81 + wi
                acc0 += v * sw[wi];
                acc1 += v * sw[81 + wi];
            }
        }
    }
    float* o = h1 + (size_t)n * 2 * HWSZ + yq * WW + xq;
    o[0]    = acc0;
    o[HWSZ] = acc1;
}

// ---------------------------------------------------------------------------
// K2: fused offset-conv (2ch -> 27ch, 3x3, pad=3 dil=3) + sigmoid(mod)
//     + modulated deformable conv (bilinear gather) -> h2 (2ch)
// Avoids materializing the 226 MB `om` tensor.
// ---------------------------------------------------------------------------
__global__ void deform_kernel(const float* __restrict__ h1,
                              const float* __restrict__ w_off,
                              const float* __restrict__ w_dcn,
                              const float* __restrict__ b_dcn,
                              float* __restrict__ h2) {
    __shared__ float swo[486];   // (27,2,3,3)
    __shared__ float swd[36];    // (2,2,3,3)
    __shared__ float sbd[2];
    int t = threadIdx.x;
    for (int i = t; i < 486; i += blockDim.x) swo[i] = w_off[i];
    if (t < 36) swd[t] = w_dcn[t];
    if (t < 2)  sbd[t] = b_dcn[t];
    __syncthreads();

    int idx = blockIdx.x * blockDim.x + t;
    if (idx >= NB * HWSZ) return;
    int xq = idx & (WW - 1);
    int yq = (idx >> 7) & (HH - 1);
    int n  = idx >> 14;

    const float* p0 = h1 + (size_t)n * 2 * HWSZ;
    const float* p1 = p0 + HWSZ;

    // dilated 3x3 taps of both channels
    float taps0[9], taps1[9];
#pragma unroll
    for (int ky = 0; ky < 3; ++ky) {
        int yy = yq + (ky - 1) * 3;
        bool oky = (yy >= 0) && (yy < HH);
#pragma unroll
        for (int kx = 0; kx < 3; ++kx) {
            int xx = xq + (kx - 1) * 3;
            bool ok = oky && (xx >= 0) && (xx < WW);
            int k = ky * 3 + kx;
            taps0[k] = ok ? p0[yy * WW + xx] : 0.f;
            taps1[k] = ok ? p1[yy * WW + xx] : 0.f;
        }
    }

    // 27 offset-conv outputs
    float om[27];
#pragma unroll
    for (int o = 0; o < 27; ++o) {
        float s = 0.f;
#pragma unroll
        for (int k = 0; k < 9; ++k) {
            s += taps0[k] * swo[o * 18 + k];
            s += taps1[k] * swo[o * 18 + 9 + k];
        }
        om[o] = s;
    }

    float acc0 = sbd[0], acc1 = sbd[1];
#pragma unroll
    for (int k = 0; k < 9; ++k) {
        // offset = concat(ox=om[0:9], oy=om[9:18]); dy=offset[2k], dx=offset[2k+1]
        float dy = om[2 * k];
        float dx = om[2 * k + 1];
        float mod = 1.f / (1.f + __expf(-om[18 + k]));
        float py = dy + (float)(((k / 3) - 1) * 3) + (float)yq;
        float px = dx + (float)(((k % 3) - 1) * 3) + (float)xq;
        float y0f = floorf(py), x0f = floorf(px);
        int   y0 = (int)y0f,   x0 = (int)x0f;
        float ly = py - y0f,   lx = px - x0f;
        float cw[4] = { (1.f - ly) * (1.f - lx), (1.f - ly) * lx,
                        ly * (1.f - lx),         ly * lx };
        int   cy[4] = { y0, y0, y0 + 1, y0 + 1 };
        int   cx[4] = { x0, x0 + 1, x0, x0 + 1 };
        float s0 = 0.f, s1 = 0.f;
#pragma unroll
        for (int c = 0; c < 4; ++c) {
            if (cy[c] >= 0 && cy[c] < HH && cx[c] >= 0 && cx[c] < WW) {
                int o = cy[c] * WW + cx[c];
                s0 += cw[c] * p0[o];
                s1 += cw[c] * p1[o];
            }
        }
        s0 *= mod; s1 *= mod;
        acc0 += s0 * swd[k]      + s1 * swd[9 + k];       // oc=0: (0,c,k)
        acc1 += s0 * swd[18 + k] + s1 * swd[27 + k];      // oc=1
    }
    float* o = h2 + (size_t)n * 2 * HWSZ + yq * WW + xq;
    o[0]    = acc0;
    o[HWSZ] = acc1;
}

// ---------------------------------------------------------------------------
// K3: 1x1 conv + relu + 2x2 maxpool + flatten, emitting bf16 activations
// one thread per pooled element (n, c, i, k)
// ---------------------------------------------------------------------------
__global__ void pool_kernel(const float* __restrict__ h2,
                            const float* __restrict__ w2,
                            const float* __restrict__ b2,
                            unsigned short* __restrict__ pooled) {
    int idx = blockIdx.x * blockDim.x + threadIdx.x;
    if (idx >= NB * 2 * 64 * 64) return;
    int kq = idx & 63;
    int iq = (idx >> 6) & 63;
    int c  = (idx >> 12) & 1;
    int n  = idx >> 13;

    float wa = w2[c * 2 + 0], wb = w2[c * 2 + 1], bb = b2[c];
    const float* p0 = h2 + (size_t)n * 2 * HWSZ;
    const float* p1 = p0 + HWSZ;
    float mx = 0.f;                               // relu outputs are >= 0
#pragma unroll
    for (int j = 0; j < 2; ++j)
#pragma unroll
        for (int l = 0; l < 2; ++l) {
            int o = (2 * iq + j) * WW + (2 * kq + l);
            float v = fmaxf(wa * p0[o] + wb * p1[o] + bb, 0.f);
            mx = fmaxf(mx, v);
        }
    pooled[(size_t)n * 8192 + c * 4096 + iq * 64 + kq] = f2bf(mx);
}

// ---------------------------------------------------------------------------
// K4: bf16 WMMA GEMM:  Out[M,Nc] = relu?(A[M,K] * W[Nc,K]^T + bias)
// M = 128 fixed. Block = 4 waves; block owns a 16-col N-tile (weights
// streamed from HBM exactly once); wave w owns M-tiles 2w, 2w+1.
// ---------------------------------------------------------------------------
__global__ void gemm_bf16_kernel(const unsigned short* __restrict__ A,
                                 const unsigned short* __restrict__ Wt,
                                 const float* __restrict__ bias,
                                 unsigned short* __restrict__ Out,
                                 int K, int Nc, int do_relu) {
    const int lane  = threadIdx.x & 31;
    const int wave  = threadIdx.x >> 5;
    const int ncol0 = blockIdx.x * 16;
    const int nr    = lane & 15;          // row-within-fragment (M or N index)
    const int kh    = (lane >> 4) * 8;    // half-wave K offset

    const unsigned short* wrow  = Wt + (size_t)(ncol0 + nr) * K;
    const unsigned short* arow0 = A + (size_t)((wave * 2 + 0) * 16 + nr) * K;
    const unsigned short* arow1 = A + (size_t)((wave * 2 + 1) * 16 + nr) * K;

    f32x8 acc0 = {}, acc1 = {};
    for (int k0 = 0; k0 < K; k0 += 32) {
        int kb = k0 + kh;
        BfFrag fb, fa0, fa1;
        fb.u[0]  = *(const u32x4*)(wrow + kb);
        fb.u[1]  = *(const u32x4*)(wrow + kb + 16);
        fa0.u[0] = *(const u32x4*)(arow0 + kb);
        fa0.u[1] = *(const u32x4*)(arow0 + kb + 16);
        fa1.u[0] = *(const u32x4*)(arow1 + kb);
        fa1.u[1] = *(const u32x4*)(arow1 + kb + 16);
        acc0 = __builtin_amdgcn_wmma_f32_16x16x32_bf16(
                   false, fa0.v, false, fb.v, (short)0, acc0, false, false);
        acc1 = __builtin_amdgcn_wmma_f32_16x16x32_bf16(
                   false, fa1.v, false, fb.v, (short)0, acc1, false, false);
    }

    // D layout: element (m = r + 8*(lane>>4), n = lane&15) lives in VGPR r
    int   nOut = ncol0 + nr;
    float bv   = bias[nOut];
    int   mb0  = (wave * 2 + 0) * 16 + (lane >> 4) * 8;
    int   mb1  = (wave * 2 + 1) * 16 + (lane >> 4) * 8;
#pragma unroll
    for (int r = 0; r < 8; ++r) {
        float v0 = acc0[r] + bv;
        float v1 = acc1[r] + bv;
        if (do_relu) { v0 = fmaxf(v0, 0.f); v1 = fmaxf(v1, 0.f); }
        Out[(size_t)(mb0 + r) * Nc + nOut] = f2bf(v0);
        Out[(size_t)(mb1 + r) * Nc + nOut] = f2bf(v1);
    }
}

// ---------------------------------------------------------------------------
// K5: final 1024 -> 1 projection + sigmoid (tiny)
// ---------------------------------------------------------------------------
__global__ void io_kernel(const unsigned short* __restrict__ a2,
                          const float* __restrict__ io_w,
                          const float* __restrict__ io_b,
                          float* __restrict__ out) {
    int n = threadIdx.x;                  // 128 threads, one per batch row
    const unsigned short* row = a2 + (size_t)n * 1024;
    float s = 0.f;
    for (int k = 0; k < 1024; ++k) s += bf2f(row[k]) * io_w[k];
    s += io_b[0];
    out[n] = 1.f / (1.f + __expf(-s));
}

// ---------------------------------------------------------------------------
// launch
// ---------------------------------------------------------------------------
extern "C" void kernel_launch(void* const* d_in, const int* in_sizes, int n_in,
                              void* d_out, int out_size, void* d_ws, size_t ws_size,
                              hipStream_t stream) {
    const float* mask  = (const float*)d_in[0];
    const float* x     = (const float*)d_in[1];
    const float* w1    = (const float*)d_in[2];
    const float* b1    = (const float*)d_in[3];
    const float* w_off = (const float*)d_in[4];
    const float* w_dcn = (const float*)d_in[5];
    const float* b_dcn = (const float*)d_in[6];
    const float* w2    = (const float*)d_in[7];
    const float* b2    = (const float*)d_in[8];
    const float* fc1_w = (const float*)d_in[9];
    const float* fc1_b = (const float*)d_in[10];
    const float* fc2_w = (const float*)d_in[11];
    const float* fc2_b = (const float*)d_in[12];
    const float* io_w  = (const float*)d_in[13];
    const float* io_b  = (const float*)d_in[14];
    float* out = (float*)d_out;

    char* ws = (char*)d_ws;
    size_t off = 0;
    auto carve = [&](size_t bytes) {
        size_t p = off; off += (bytes + 255) & ~(size_t)255; return p;
    };
    float*          h1     = (float*)         (ws + carve((size_t)NB * 2 * HWSZ * 4));
    float*          h2     = (float*)         (ws + carve((size_t)NB * 2 * HWSZ * 4));
    unsigned short* pooled = (unsigned short*)(ws + carve((size_t)NB * 8192 * 2));
    unsigned short* w1bf   = (unsigned short*)(ws + carve((size_t)2048 * 8192 * 2));
    unsigned short* w2bf   = (unsigned short*)(ws + carve((size_t)1024 * 2048 * 2));
    unsigned short* a1     = (unsigned short*)(ws + carve((size_t)NB * 2048 * 2));
    unsigned short* a2     = (unsigned short*)(ws + carve((size_t)NB * 1024 * 2));

    // weight conversion (bandwidth-only; removes 8x WMMA instruction count vs f32 path)
    cvt_bf16_kernel<<<4096, 256, 0, stream>>>(fc1_w, w1bf, 2048 * 8192);
    cvt_bf16_kernel<<<2048, 256, 0, stream>>>(fc2_w, w2bf, 1024 * 2048);

    // conv pipeline
    conv1_kernel <<<(NB * HWSZ) / 256, 256, 0, stream>>>(x, mask, w1, b1, h1);
    deform_kernel<<<(NB * HWSZ) / 256, 256, 0, stream>>>(h1, w_off, w_dcn, b_dcn, h2);
    pool_kernel  <<<(NB * 2 * 64 * 64) / 256, 256, 0, stream>>>(h2, w2, b2, pooled);

    // FC layers on WMMA (f32 accumulate)
    gemm_bf16_kernel<<<2048 / 16, 128, 0, stream>>>(pooled, w1bf, fc1_b, a1, 8192, 2048, 1);
    gemm_bf16_kernel<<<1024 / 16, 128, 0, stream>>>(a1,     w2bf, fc2_b, a2, 2048, 1024, 1);

    io_kernel<<<1, 128, 0, stream>>>(a2, io_w, io_b, out);

    (void)in_sizes; (void)n_in; (void)out_size; (void)ws_size;
}
